// OLinear_Q_33818572489493
// MI455X (gfx1250) — matrix operand order repaired and checked by reading, not compiled, not run on hardware
//
#include <hip/hip_runtime.h>
#include <hip/hip_bf16.h>

typedef _Float16 v16h __attribute__((ext_vector_type(16)));
typedef _Float16 v8h  __attribute__((ext_vector_type(8)));
typedef float    v8f  __attribute__((ext_vector_type(8)));

#define B_DIM 4096
#define CIN   2048
#define COUT  2048

// ---- workspace layout (bytes) ----
#define WS_TX   256
#define WS_TW   16640
#define WS_XH   24832
#define WS_WH   (24832 + 16777216)

union Frag { v16h v; v8h h[2]; };

__global__ void k_init(unsigned int* maxbits) {
    if (threadIdx.x == 0) *maxbits = 0u;
}

__global__ void k_absmax(const float* __restrict__ w, unsigned int* __restrict__ maxbits) {
    __shared__ float red[256];
    float m = 0.f;
    for (int i = blockIdx.x * 256 + threadIdx.x; i < COUT * CIN; i += gridDim.x * 256)
        m = fmaxf(m, fabsf(w[i]));
    red[threadIdx.x] = m;
    __syncthreads();
    for (int s = 128; s > 0; s >>= 1) {
        if (threadIdx.x < s) red[threadIdx.x] = fmaxf(red[threadIdx.x], red[threadIdx.x + s]);
        __syncthreads();
    }
    if (threadIdx.x == 0) atomicMax(maxbits, __float_as_uint(red[0]));
}

__global__ void k_quant_w(const float* __restrict__ w, const float* __restrict__ phases,
                          const float* __restrict__ disks, const unsigned int* __restrict__ maxbits,
                          _Float16* __restrict__ Wh, float* __restrict__ term_w) {
    __shared__ float red[256];
    const int o = blockIdx.x;
    const float maxt = tanhf(__uint_as_float(*maxbits));   // max|tanh(w)| = tanh(max|w|)
    const float inv2 = 0.5f / maxt;
    float acc = 0.f;
#pragma unroll
    for (int j = 0; j < CIN / 256; j++) {
        const int i = threadIdx.x + j * 256;
        const float t = tanhf(w[o * CIN + i]);
        const float v = t * inv2 + 0.5f;
        const float q = rintf(v * 255.f) * (1.f / 255.f);
        const float d0 = disks[2 * i], d1 = disks[2 * i + 1];
        const float s = sinf(phases[i]) * (d0 + d1);
        Wh[o * CIN + i] = (_Float16)(q * s);
        acc += q * q * (d0 - d1);
    }
    red[threadIdx.x] = acc;
    __syncthreads();
    for (int s = 128; s > 0; s >>= 1) {
        if (threadIdx.x < s) red[threadIdx.x] += red[threadIdx.x + s];
        __syncthreads();
    }
    if (threadIdx.x == 0) term_w[o] = red[0];
}

__global__ void k_quant_x(const float* __restrict__ x, const float* __restrict__ disks,
                          _Float16* __restrict__ Xh, float* __restrict__ term_x) {
    __shared__ float red[256];
    const int b = blockIdx.x;
    float acc = 0.f;
#pragma unroll
    for (int j = 0; j < CIN / 256; j++) {
        const int i = threadIdx.x + j * 256;
        float v = x[b * CIN + i];
        v = fminf(fmaxf(v, 0.f), 1.f);
        const float q = rintf(v * 255.f) * (1.f / 255.f);
        Xh[b * CIN + i] = (_Float16)q;
        const float d0 = disks[2 * i], d1 = disks[2 * i + 1];
        acc += q * q * (d0 - d1);
    }
    red[threadIdx.x] = acc;
    __syncthreads();
    for (int s = 128; s > 0; s >>= 1) {
        if (threadIdx.x < s) red[threadIdx.x] += red[threadIdx.x + s];
        __syncthreads();
    }
    if (threadIdx.x == 0) term_x[b] = red[0];
}

// ---- GEMM: out[b,o] = 0.25*term_x[b] + 0.25*term_w[o] + 0.5 * (Xh @ Wh^T)[b,o] ----
#define BM 128
#define BN 128
#define BK 64
#define LSTR 72   // LDS row stride in halves (144B rows: 16B-aligned, conflict-avoiding)

__global__ __launch_bounds__(256, 2)
void k_gemm(const _Float16* __restrict__ Xh, const _Float16* __restrict__ Wh,
            const float* __restrict__ term_x, const float* __restrict__ term_w,
            float* __restrict__ out) {
    __shared__ _Float16 Xs[2][BM * LSTR];
    __shared__ _Float16 Ws[2][BN * LSTR];

    const int tid   = threadIdx.x;
    const int lane  = tid & 31;
    const int wave  = tid >> 5;
    const int wavem = wave & 3;   // 4 waves along M (32 rows each)
    const int waven = wave >> 2;  // 2 waves along N (64 cols each)
    const int blkN  = blockIdx.x * BN;
    const int blkM  = blockIdx.y * BM;

    // Async global->LDS staging (CDNA5 path, tracked by ASYNCcnt):
    // 1024 16B chunks per 128x64-half tile, 4 chunks/thread per array.
    auto issue_tile = [&](int kt, int buf) {
#pragma unroll
        for (int j = 0; j < 4; j++) {
            const int c   = tid + j * 256;
            const int r   = c >> 3;
            const int col = (c & 7) << 3;
            const unsigned xlds = (unsigned)(size_t)&Xs[buf][r * LSTR + col];
            const unsigned wlds = (unsigned)(size_t)&Ws[buf][r * LSTR + col];
            const unsigned long long xg =
                (unsigned long long)(size_t)&Xh[(size_t)(blkM + r) * CIN + kt + col];
            const unsigned long long wg =
                (unsigned long long)(size_t)&Wh[(size_t)(blkN + r) * CIN + kt + col];
            asm volatile("global_load_async_to_lds_b128 %0, %1, off"
                         :: "v"(xlds), "v"(xg) : "memory");
            asm volatile("global_load_async_to_lds_b128 %0, %1, off"
                         :: "v"(wlds), "v"(wg) : "memory");
        }
    };

    const v8f zero = {0.f, 0.f, 0.f, 0.f, 0.f, 0.f, 0.f, 0.f};
    v8f acc[2][4];
#pragma unroll
    for (int i = 0; i < 2; i++)
#pragma unroll
        for (int j = 0; j < 4; j++) acc[i][j] = zero;

    // prologue: stage first tile
    issue_tile(0, 0);
    asm volatile("s_wait_asynccnt 0x0" ::: "memory");
    __syncthreads();

    int cur = 0;
    for (int kt = 0; kt < CIN; kt += BK) {
        const bool more = (kt + BK < CIN);
        if (more) issue_tile(kt + BK, cur ^ 1);  // overlap next tile with compute

#pragma unroll
        for (int kc = 0; kc < BK; kc += 32) {
            // A: lane holds row (lane&15); halves 0..7 -> K=base..base+7,
            //    halves 8..15 -> K=base+16..base+23, base = (lane<16)?0:8
            const int abase = (lane < 16) ? 0 : 8;
            Frag a[2];
#pragma unroll
            for (int sm = 0; sm < 2; sm++) {
                const _Float16* p =
                    &Xs[cur][(wavem * 32 + sm * 16 + (lane & 15)) * LSTR + kc + abase];
                a[sm].h[0] = *(const v8h*)(p);
                a[sm].h[1] = *(const v8h*)(p + 16);
            }
            // B: lane holds column (lane&15); halves 0..7 -> K=base..base+7,
            //    halves 8..15 -> K=base+16..base+23, base = (lane<16)?0:8
            const int bbase = (lane < 16) ? 0 : 8;
            Frag b[4];
#pragma unroll
            for (int sn = 0; sn < 4; sn++) {
                const _Float16* p =
                    &Ws[cur][(waven * 64 + sn * 16 + (lane & 15)) * LSTR + kc + bbase];
                b[sn].h[0] = *(const v8h*)(p);
                b[sn].h[1] = *(const v8h*)(p + 16);
            }
#pragma unroll
            for (int sn = 0; sn < 4; sn++)
#pragma unroll
                for (int sm = 0; sm < 2; sm++)
                    acc[sm][sn] = __builtin_amdgcn_wmma_f32_16x16x32_f16(
                        false, a[sm].v, false, b[sn].v, (short)0, acc[sm][sn], false, false);
        }

        if (more) {
            asm volatile("s_wait_asynccnt 0x0" ::: "memory");
            __syncthreads();   // everyone's async data landed; safe to flip buffers
        }
        cur ^= 1;
    }

    // writeback: D VGPR r -> row r + 8*(lane>=16); col = lane&15
    const int mrow0 = blkM + wavem * 32 + ((lane >= 16) ? 8 : 0);
#pragma unroll
    for (int sm = 0; sm < 2; sm++) {
#pragma unroll
        for (int sn = 0; sn < 4; sn++) {
            const int n  = blkN + waven * 64 + sn * 16 + (lane & 15);
            const float tw = 0.25f * term_w[n];
#pragma unroll
            for (int r = 0; r < 8; r++) {
                const int m = mrow0 + sm * 16 + r;
                out[m * COUT + n] = 0.25f * term_x[m] + tw + 0.5f * acc[sm][sn][r];
            }
        }
    }
}

extern "C" void kernel_launch(void* const* d_in, const int* in_sizes, int n_in,
                              void* d_out, int out_size, void* d_ws, size_t ws_size,
                              hipStream_t stream) {
    const float* x      = (const float*)d_in[0];   // (4096, 2048)
    const float* weight = (const float*)d_in[1];   // (2048, 2048)
    const float* phases = (const float*)d_in[2];   // (2048,)
    const float* disks  = (const float*)d_in[3];   // (2048, 2)
    float* out = (float*)d_out;

    char* ws = (char*)d_ws;
    unsigned int* maxbits = (unsigned int*)(ws);
    float*     term_x = (float*)(ws + WS_TX);
    float*     term_w = (float*)(ws + WS_TW);
    _Float16*  Xh     = (_Float16*)(ws + WS_XH);
    _Float16*  Wh     = (_Float16*)(ws + WS_WH);

    k_init<<<1, 1, 0, stream>>>(maxbits);
    k_absmax<<<1024, 256, 0, stream>>>(weight, maxbits);
    k_quant_w<<<COUT, 256, 0, stream>>>(weight, phases, disks, maxbits, Wh, term_w);
    k_quant_x<<<B_DIM, 256, 0, stream>>>(x, disks, Xh, term_x);

    dim3 grid(COUT / BN, B_DIM / BM);  // 16 x 32 blocks
    k_gemm<<<grid, 256, 0, stream>>>(Xh, Wh, term_x, term_w, out);
}
